// CrossAttentionTemporal_76227079569557
// MI455X (gfx1250) — matrix operand order repaired and checked by reading, not compiled, not run on hardware
//
#include <hip/hip_runtime.h>

// ---------------------------------------------------------------------------
// Fused temporal cross-attention for MI455X (gfx1250, wave32, WMMA).
//   x:[2,2048,16,512] f32 -> out:[2,2048,16,512] f32
// One fused kernel; weights pre-converted to WMMA-swizzled f16 in d_ws.
// ---------------------------------------------------------------------------

typedef _Float16 f16;
typedef __attribute__((ext_vector_type(16))) _Float16 v16h;
typedef __attribute__((ext_vector_type(8)))  _Float16 v8h;
typedef __attribute__((ext_vector_type(4)))  _Float16 v4h;
typedef __attribute__((ext_vector_type(8)))  float    v8f;

#define HEADS     8
#define DHEAD     64
#define CDIM      512          // QUERY_DIM == INNER
#define FR        16           // frames per slice
#define WG_ROWS   64           // 4 slices of 16 frames per workgroup
#define STR       520          // padded LDS row stride (f16 elements)
#define NROWS_ALL 65536        // b*s*f
#define WMAT_ELEMS (CDIM*CDIM) // 262144 elements per weight matrix

// LDS plan (bytes):
//   Xh (reused as Oh) : 64*520*2 = 66560
//   Qh, Kh, Vh        : 3 * 66560
//   pos_bias f32      : 8*16*16*4 = 8192
//   per-wave P scratch: 8 * 16*32*2 = 8192
#define XSZ        (WG_ROWS * STR)             // f16 elements per matrix buf
#define SMEM_BYTES (4 * XSZ * 2 + 8192 + 8192) // 282624 <= 320KB

__device__ __forceinline__ v8f wmma16(v16h a, v16h b, v8f c) {
  // D = A(16x32 f16) * B(32x16 f16) + C(16x16 f32)
  return __builtin_amdgcn_wmma_f32_16x16x32_f16(
      false, a, false, b, (short)0, c, false, false);
}

// A-fragment (16x32, row-major source; lanes index M).
// Layout: element s of lane half h holds k = half*16 + s (contiguous).
__device__ __forceinline__ v16h ldsA(const f16* p, int stride, int row0, int col0) {
  const int lane = threadIdx.x & 31;
  const int hf   = lane >> 4;
  const f16* q = p + (row0 + (lane & 15)) * stride + col0 + hf * 16;
  v8h a = *(const v8h*)(q);
  v8h b = *(const v8h*)(q + 8);
  v16h r;
#pragma unroll
  for (int i = 0; i < 8; ++i) { r[i] = a[i]; r[i + 8] = b[i]; }
  return r;
}

// B-fragment of a TRANSPOSED source: B[k][n] = src[n][k] (lanes index N).
// B layout used here: lanes 0-15 hold k=0..15, lanes 16-31 hold k=16..31 (contiguous).
__device__ __forceinline__ v16h ldsBT(const f16* p, int stride, int row0, int col0) {
  const int lane = threadIdx.x & 31;
  const int hf   = lane >> 4;
  const f16* q = p + (row0 + (lane & 15)) * stride + col0 + hf * 16;
  v8h a = *(const v8h*)(q);
  v8h b = *(const v8h*)(q + 8);
  v16h r;
#pragma unroll
  for (int i = 0; i < 8; ++i) { r[i] = a[i]; r[i + 8] = b[i]; }
  return r;
}

// B-fragment from pre-swizzled weight block (one contiguous 32B per lane).
__device__ __forceinline__ v16h wsB(const f16* blockBase) {
  const int lane = threadIdx.x & 31;
  return *(const v16h*)(blockBase + lane * 16);
}

// ---------------------------------------------------------------------------
// Weight converter: fp32 [512][512] row-major -> f16 B-fragment-swizzled.
// Block (ntile,kstep) of 32x16 stored as [lane][16 f16] contiguous (1KB).
// ---------------------------------------------------------------------------
__global__ __launch_bounds__(256) void wconv_kernel(
    const float* __restrict__ Wq, const float* __restrict__ Wk,
    const float* __restrict__ Wv, const float* __restrict__ Wo,
    f16* __restrict__ wsW) {
  int idx = blockIdx.x * 256 + threadIdx.x;       // 4 * 262144 total
  int mat = idx >> 18;
  int rem = idx & (WMAT_ELEMS - 1);
  int k = rem >> 9, n = rem & 511;
  const float* W = (mat == 0) ? Wq : (mat == 1) ? Wk : (mat == 2) ? Wv : Wo;
  float val = W[k * CDIM + n];
  int ntile = n >> 4, nn = n & 15;
  int kstep = k >> 5, kk = k & 31;
  int hf = kk >> 4, slot = kk & 15;               // B layout: k = half*16 + slot
  int lane = nn + 16 * hf;
  wsW[(size_t)mat * WMAT_ELEMS + (ntile * 16 + kstep) * 512 + lane * 16 + slot] =
      (f16)val;
}

// ---------------------------------------------------------------------------
// Fused kernel: 1024 blocks x 256 threads, 64 rows (4 slices) per block.
// ---------------------------------------------------------------------------
__global__ __launch_bounds__(256) void fused_attn_kernel(
    const float* __restrict__ x, const float* __restrict__ pos_bias,
    const float* __restrict__ bo, const f16* __restrict__ wsW,
    float* __restrict__ out) {
  extern __shared__ char smemraw[];
  f16*   Xh    = (f16*)smemraw;            // reused as attention-output Oh
  f16*   Qh    = Xh + XSZ;
  f16*   Kh    = Qh + XSZ;
  f16*   Vh    = Kh + XSZ;
  float* biasS = (float*)(Vh + XSZ);       // 2048 f32
  f16*   pscr  = (f16*)(biasS + 2048);     // 8 waves * 16*32 f16

  const int tid  = threadIdx.x;
  const int wave = tid >> 5;
  const int lane = tid & 31;
  const int hf   = lane >> 4;
  const int l15  = lane & 15;
  const size_t r0 = (size_t)blockIdx.x * WG_ROWS;

  // ---- Phase 1: stage X (f32 -> f16) and pos_bias into LDS ----------------
#pragma unroll
  for (int i = 0; i < 32; ++i) {                  // 8192 float4 total
    int fi  = tid + i * 256;
    int row = fi >> 7;
    int c4  = (fi & 127) * 4;
    float4 vx = *(const float4*)(x + (r0 + row) * CDIM + c4);
    v4h hv = { (f16)vx.x, (f16)vx.y, (f16)vx.z, (f16)vx.w };
    *(v4h*)(Xh + row * STR + c4) = hv;
  }
#pragma unroll
  for (int i = 0; i < 8; ++i) biasS[tid * 8 + i] = pos_bias[tid * 8 + i];

  // Zero the k=16..31 pad region of this wave's P scratch (16x32 layout).
  f16* ws0 = pscr + wave * 512;
  { v8h z = {}; *(v8h*)(ws0 + l15 * 32 + 16 + hf * 8) = z; }

  __syncthreads();

  // ---- Phase 2: Q/K/V projection GEMMs (WMMA f16 -> f32 acc) --------------
  // Each wave owns ntiles [wave*4, wave*4+4) for every (matrix, mtile).
#pragma unroll
  for (int mat = 0; mat < 3; ++mat) {
    f16* dst = (mat == 0) ? Qh : (mat == 1) ? Kh : Vh;
    const float scl = (mat == 0) ? 0.125f : 1.0f;  // DIM_HEAD^-0.5 on Q
    const f16* WB = wsW + (size_t)mat * WMAT_ELEMS;
    for (int mtile = 0; mtile < 4; ++mtile) {
      v8f acc[4] = {};
      for (int ks = 0; ks < 16; ++ks) {
        v16h a = ldsA(Xh, STR, mtile * 16, ks * 32);
        if (ks + 1 < 16)
          __builtin_prefetch(WB + ((wave * 4) * 16 + ks + 1) * 512, 0, 1);
#pragma unroll
        for (int nt = 0; nt < 4; ++nt) {
          v16h b = wsB(WB + ((wave * 4 + nt) * 16 + ks) * 512);
          acc[nt] = wmma16(a, b, acc[nt]);
        }
      }
#pragma unroll
      for (int nt = 0; nt < 4; ++nt) {
        int ntile = wave * 4 + nt;
#pragma unroll
        for (int v = 0; v < 8; ++v) {
          int m = v + 8 * hf;                      // C-fragment row
          dst[(mtile * 16 + m) * STR + ntile * 16 + l15] = (f16)(acc[nt][v] * scl);
        }
      }
    }
  }
  __syncthreads();

  // ---- Phase 3: per-(slice,head) 16x16 causal attention -------------------
  for (int pair = wave; pair < 32; pair += 8) {
    int g = pair >> 3, h = pair & 7;
    int rb = g * 16, cb = h * DHEAD;

    // scores S = Qs * K^T  (16x16, K-dim 64 = 2 WMMA)
    v8f s = {};
#pragma unroll
    for (int ks = 0; ks < 2; ++ks) {
      v16h a = ldsA (Qh, STR, rb, cb + ks * 32);
      v16h b = ldsBT(Kh, STR, rb, cb + ks * 32);
      s = wmma16(a, b, s);
    }

    // + pos_bias, strict causal mask, softmax (rows live across 16 lanes)
    float p[8];
#pragma unroll
    for (int v = 0; v < 8; ++v) {
      int m = v + 8 * hf;
      float sv = s[v] + biasS[h * 256 + m * 16 + l15];
      if (l15 > m) sv = -1e30f;                    // triu(k=1) mask
      p[v] = sv;
    }
#pragma unroll
    for (int v = 0; v < 8; ++v) {
      float r = p[v];
#pragma unroll
      for (int msk = 1; msk < 16; msk <<= 1) r = fmaxf(r, __shfl_xor(r, msk, 32));
      float e = __expf(p[v] - r);                  // T_TEMP == 1
      float sm = e;
#pragma unroll
      for (int msk = 1; msk < 16; msk <<= 1) sm += __shfl_xor(sm, msk, 32);
      p[v] = e / sm;
    }

    // stage P (16x16, zero-padded to K=32) for A-fragment reload
#pragma unroll
    for (int v = 0; v < 8; ++v) ws0[(v + 8 * hf) * 32 + l15] = (f16)p[v];

    // O_h = P @ V_h   (16x64 = 4 WMMA, K padded 16->32)
    v16h a = ldsA(ws0, 32, 0, 0);
#pragma unroll
    for (int nt = 0; nt < 4; ++nt) {
      v16h b = {};
      if (hf == 0) {                               // lanes 16-31 carry k>=16 (pad=0)
#pragma unroll
        for (int kk = 0; kk < 16; ++kk)
          b[kk] = Vh[(rb + kk) * STR + cb + nt * 16 + l15];
      }
      v8f o = {};
      o = wmma16(a, b, o);
#pragma unroll
      for (int v = 0; v < 8; ++v) {
        int m = v + 8 * hf;
        Xh[(rb + m) * STR + cb + nt * 16 + l15] = (f16)o[v];  // Oh reuses Xh
      }
    }
  }
  __syncthreads();

  // ---- Phase 4: output GEMM  out = Oh @ Wo + bo ---------------------------
  const f16* WB = wsW + (size_t)3 * WMAT_ELEMS;
  for (int mtile = 0; mtile < 4; ++mtile) {
    v8f acc[4] = {};
    for (int ks = 0; ks < 16; ++ks) {
      v16h a = ldsA(Xh, STR, mtile * 16, ks * 32);
      if (ks + 1 < 16)
        __builtin_prefetch(WB + ((wave * 4) * 16 + ks + 1) * 512, 0, 1);
#pragma unroll
      for (int nt = 0; nt < 4; ++nt) {
        v16h b = wsB(WB + ((wave * 4 + nt) * 16 + ks) * 512);
        acc[nt] = wmma16(a, b, acc[nt]);
      }
    }
#pragma unroll
    for (int nt = 0; nt < 4; ++nt) {
      int ntile = wave * 4 + nt;
      float bias = bo[ntile * 16 + l15];
#pragma unroll
      for (int v = 0; v < 8; ++v) {
        int m = v + 8 * hf;
        out[(r0 + mtile * 16 + m) * CDIM + ntile * 16 + l15] = acc[nt][v] + bias;
      }
    }
  }
}

// ---------------------------------------------------------------------------
extern "C" void kernel_launch(void* const* d_in, const int* in_sizes, int n_in,
                              void* d_out, int out_size, void* d_ws, size_t ws_size,
                              hipStream_t stream) {
  const float* x  = (const float*)d_in[0];
  const float* pb = (const float*)d_in[1];
  const float* Wq = (const float*)d_in[2];
  const float* Wk = (const float*)d_in[3];
  const float* Wv = (const float*)d_in[4];
  const float* Wo = (const float*)d_in[5];
  const float* bo = (const float*)d_in[6];
  float* out = (float*)d_out;
  f16*   wsW = (f16*)d_ws;                         // 4 * 512KB f16 weights

  // 1) swizzle+convert weights into d_ws (L2-resident, rebuilt every call)
  wconv_kernel<<<(4 * WMAT_ELEMS) / 256, 256, 0, stream>>>(Wq, Wk, Wv, Wo, wsW);

  // 2) fused QKV + attention + output projection
  (void)hipFuncSetAttribute((const void*)fused_attn_kernel,
                            hipFuncAttributeMaxDynamicSharedMemorySize,
                            (int)SMEM_BYTES);
  fused_attn_kernel<<<NROWS_ALL / WG_ROWS, 256, SMEM_BYTES, stream>>>(
      x, pb, bo, wsW, out);
}
